// GIN_69097433858367
// MI455X (gfx1250) — matrix operand order repaired and checked by reading, not compiled, not run on hardware
//
#include <hip/hip_runtime.h>

#define N_FEAT 256
#define N_HID  512
#define N_CLS  256

typedef __attribute__((ext_vector_type(16))) __bf16 v16bf;
typedef __attribute__((ext_vector_type(8)))  float  v8f;
typedef __attribute__((ext_vector_type(4)))  unsigned int v4u;
typedef __attribute__((ext_vector_type(8)))  int  v8i;
typedef __attribute__((ext_vector_type(4)))  int  v4i;

// ---------------- weight convert + transpose: W[K][N] f32 -> Wt[N][K] bf16 --
__global__ void wconv_kernel(const float* __restrict__ W, __bf16* __restrict__ Wt,
                             int K, int N) {
  int idx = blockIdx.x * blockDim.x + threadIdx.x;
  if (idx >= K * N) return;
  int k = idx / N, n = idx % N;
  Wt[(size_t)n * K + k] = (__bf16)W[idx];
}

// ---------------- self term: out = (1+eps)*x ------------------------------
__global__ void scale_kernel(const float* __restrict__ x, const float* __restrict__ eps,
                             float* __restrict__ out, long long n) {
  long long i = (long long)blockIdx.x * blockDim.x + threadIdx.x;
  if (i < n) out[i] = (1.0f + eps[0]) * x[i];
}

// ---------------- edge scatter: agg[dst] += h[src] ------------------------
template <int F>
__global__ void scatter_kernel(const float* __restrict__ h, const int* __restrict__ ei,
                               float* __restrict__ agg, int n_edges) {
  const int G = F / 4;
  long long t = (long long)blockIdx.x * blockDim.x + threadIdx.x;
  int e = (int)(t / G);
  if (e >= n_edges) return;
  int c = (int)(t % G) * 4;
  int src = ei[e];
  int dst = ei[n_edges + e];
  const float4 v = *(const float4*)(h + (size_t)src * F + c);
  float* d = agg + (size_t)dst * F + c;
  atomicAdd(d + 0, v.x);
  atomicAdd(d + 1, v.y);
  atomicAdd(d + 2, v.z);
  atomicAdd(d + 3, v.w);
}

__device__ __forceinline__ v16bf cvt_frag(const float4* r) {
  v16bf f;
  f[0]  = (__bf16)r[0].x; f[1]  = (__bf16)r[0].y; f[2]  = (__bf16)r[0].z; f[3]  = (__bf16)r[0].w;
  f[4]  = (__bf16)r[1].x; f[5]  = (__bf16)r[1].y; f[6]  = (__bf16)r[1].z; f[7]  = (__bf16)r[1].w;
  f[8]  = (__bf16)r[2].x; f[9]  = (__bf16)r[2].y; f[10] = (__bf16)r[2].z; f[11] = (__bf16)r[2].w;
  f[12] = (__bf16)r[3].x; f[13] = (__bf16)r[3].y; f[14] = (__bf16)r[3].z; f[15] = (__bf16)r[3].w;
  return f;
}

// ---------------- WMMA GEMM: C = act(A @ W + b) ---------------------------
// A: M x K f32 (row-major). Wt: NOUT x K bf16 (transposed weights).
// Block = 256 threads = 8 waves, computes 256 rows x 128 cols of C
// (each wave: 32 rows x 128 cols = 16 WMMA tiles, so every B fragment
// loaded from LDS feeds two WMMAs). Weight strip Wt[n0..n0+127][0..K) is
// staged into LDS once per block by the Tensor Data Mover (padded pitch
// K*2+32 B to spread banks). A-fragment loads are software-pipelined.
template <int K, int NOUT, bool RELU>
__global__ __launch_bounds__(256) void gemm_kernel(const float* __restrict__ A,
                                                   const __bf16* __restrict__ Wt,
                                                   const float* __restrict__ bias,
                                                   float* __restrict__ C, int M) {
  extern __shared__ char smem[];
  constexpr int PITCH = K * 2 + 32;     // bytes per LDS row (8-DWORD TDM pad)

  const int wave  = threadIdx.x >> 5;   // 0..7 -> 32-row group
  const int lane  = threadIdx.x & 31;
  const int l16   = lane & 15;
  const int lhalf = lane >> 4;          // 0: lanes 0-15, 1: lanes 16-31
  const int m0 = blockIdx.x * 256 + wave * 32;
  const int n0 = blockIdx.y * 128;

  // ---- TDM: DMA the 128 x K bf16 weight strip into LDS (one op per block)
  if (wave == 0) {
    const unsigned long long ga = (unsigned long long)(uintptr_t)(Wt + (size_t)n0 * K);
    const unsigned lds_base = (unsigned)(uintptr_t)smem;
    // D# group 0: count=1, lds_addr, global_addr[56:0], type=2 ("image")
    v4u g0 = { 1u,
               lds_base,
               (unsigned)ga,
               (unsigned)((ga >> 32) & 0x01FFFFFFu) | (2u << 30) };
    // pad after each row (K=512: 256 dwords = code 7; K=256: 128 = code 6),
    // pad_amount code 7 -> 8 dwords (32 B)
    constexpr unsigned padi = (K == 512) ? 7u : 6u;
    v8i g1;
    g1[0] = (int)((1u << 16)          // data_size = 1 -> 2 bytes
                  | (1u << 20)        // pad_enable
                  | (padi << 22)      // pad_interval
                  | (7u << 25));      // pad_amount = 8 dwords
    g1[1] = (int)(((unsigned)K & 0xFFFFu) << 16);     // tensor_dim0[15:0]
    g1[2] = (int)(((unsigned)NOUT & 0xFFFFu) << 16);  // dim0 hi=0 | tensor_dim1[15:0]
    g1[3] = (int)(((unsigned)K & 0xFFFFu) << 16);     // dim1 hi=0 | tile_dim0 = K
    g1[4] = 128;                                      // tile_dim1 = 128 rows
    g1[5] = K;                                        // tensor_dim0_stride
    g1[6] = 0;
    g1[7] = 0;
    v4i gz4 = {0, 0, 0, 0};
    v8i gz8 = {0, 0, 0, 0, 0, 0, 0, 0};
    __builtin_amdgcn_tensor_load_to_lds(g0, g1, gz4, gz4, gz8, 0);
    __builtin_amdgcn_s_wait_tensorcnt(0);
  }
  __syncthreads();

  // A-fragment rows for this lane (clamped so EXEC stays all-1s for WMMA)
  int mr0 = m0 + l16;      if (mr0 >= M) mr0 = M - 1;
  int mr1 = m0 + 16 + l16; if (mr1 >= M) mr1 = M - 1;
  // lanes 0-15 read K = k0+{0..15}; lanes 16-31 read K = k0+{16..31}
  const float* __restrict__ arow0 = A + (size_t)mr0 * K + lhalf * 16;
  const float* __restrict__ arow1 = A + (size_t)mr1 * K + lhalf * 16;

  v8f acc0[8], acc1[8];
  const v8f vzero = {0.f, 0.f, 0.f, 0.f, 0.f, 0.f, 0.f, 0.f};
#pragma unroll
  for (int t = 0; t < 8; ++t) { acc0[t] = vzero; acc1[t] = vzero; }

  // software-pipelined raw A staging (two 16-row fragments per wave)
  float4 r0[4], r1[4];
#pragma unroll
  for (int i = 0; i < 4; ++i) {
    const int o = i * 4;   // 0,4,8,12
    r0[i] = *(const float4*)(arow0 + o);
    r1[i] = *(const float4*)(arow1 + o);
  }

  for (int k0 = 0; k0 < K; k0 += 32) {
    const v16bf af0 = cvt_frag(r0);
    const v16bf af1 = cvt_frag(r1);

    // prefetch next k-step's raw A (clamped so the address is always valid)
    const int kn = (k0 + 32 < K) ? (k0 + 32) : 0;
    float4 p0[4], p1[4];
#pragma unroll
    for (int i = 0; i < 4; ++i) {
      const int o = kn + i * 4;
      p0[i] = *(const float4*)(arow0 + o);
      p1[i] = *(const float4*)(arow1 + o);
    }

    const int kbyte = (k0 + lhalf * 16) * 2;
#pragma unroll
    for (int t = 0; t < 8; ++t) {
      // B 32x16 fragment from LDS: lane column n = t*16+l16 (= LDS row),
      // lanes 0-15 carry K=k0..k0+15, lanes 16-31 carry K=k0+16..k0+31.
      const int r = t * 16 + l16;
      const v16bf bf = *(const v16bf*)(smem + (size_t)r * PITCH + kbyte);
      acc0[t] = __builtin_amdgcn_wmma_f32_16x16x32_bf16(
          false, af0, false, bf, (short)0, acc0[t], false, false);
      acc1[t] = __builtin_amdgcn_wmma_f32_16x16x32_bf16(
          false, af1, false, bf, (short)0, acc1[t], false, false);
    }

#pragma unroll
    for (int i = 0; i < 4; ++i) { r0[i] = p0[i]; r1[i] = p1[i]; }
  }

  // C/D layout: VGPR v, lanes 0-15 -> M = v; lanes 16-31 -> M = v+8; N = l16.
  if (m0 + 32 <= M) {
    // interior wave: no bounds checks (wave-uniform fast path)
#pragma unroll
    for (int t = 0; t < 8; ++t) {
      const int n = n0 + t * 16 + l16;
      const float bv = bias[n];
      float* __restrict__ cp = C + (size_t)(m0 + lhalf * 8) * NOUT + n;
#pragma unroll
      for (int v = 0; v < 8; ++v) {
        float v0 = acc0[t][v] + bv;
        float v1 = acc1[t][v] + bv;
        if (RELU) { v0 = fmaxf(v0, 0.0f); v1 = fmaxf(v1, 0.0f); }
        cp[(size_t)v * NOUT] = v0;
        cp[(size_t)(v + 16) * NOUT] = v1;
      }
    }
  } else {
    // boundary wave: per-row guards
#pragma unroll
    for (int t = 0; t < 8; ++t) {
      const int n = n0 + t * 16 + l16;
      const float bv = bias[n];
#pragma unroll
      for (int v = 0; v < 8; ++v) {
        int m = m0 + lhalf * 8 + v;
        if (m < M) {
          float val = acc0[t][v] + bv;
          if (RELU) val = fmaxf(val, 0.0f);
          C[(size_t)m * NOUT + n] = val;
        }
        m += 16;
        if (m < M) {
          float val = acc1[t][v] + bv;
          if (RELU) val = fmaxf(val, 0.0f);
          C[(size_t)m * NOUT + n] = val;
        }
      }
    }
  }
}

// ---------------------------------------------------------------------------
extern "C" void kernel_launch(void* const* d_in, const int* in_sizes, int n_in,
                              void* d_out, int out_size, void* d_ws, size_t ws_size,
                              hipStream_t stream) {
  const float* x    = (const float*)d_in[0];
  const int*   ei   = (const int*)d_in[1];
  const float* W1a  = (const float*)d_in[2];
  const float* b1a  = (const float*)d_in[3];
  const float* W1b  = (const float*)d_in[4];
  const float* b1b  = (const float*)d_in[5];
  const float* W2a  = (const float*)d_in[6];
  const float* b2a  = (const float*)d_in[7];
  const float* W2b  = (const float*)d_in[8];
  const float* b2b  = (const float*)d_in[9];
  const float* eps1 = (const float*)d_in[10];
  const float* eps2 = (const float*)d_in[11];

  const int M = in_sizes[0] / N_FEAT;   // 100000 nodes
  const int E = in_sizes[1] / 2;        // 1600000 edges

  // workspace carve-out
  char*  ws  = (char*)d_ws;
  size_t off = 0;
  auto carve = [&](size_t bytes) -> void* {
    void* p = ws + off;
    off = (off + bytes + 255) & ~(size_t)255;
    return p;
  };
  float*  bufA = (float*)carve((size_t)M * N_HID * sizeof(float));
  float*  bufB = (float*)carve((size_t)M * N_HID * sizeof(float));
  __bf16* Wt1a = (__bf16*)carve((size_t)N_FEAT * N_HID * sizeof(__bf16));
  __bf16* Wt1b = (__bf16*)carve((size_t)N_HID * N_HID * sizeof(__bf16));
  __bf16* Wt2a = (__bf16*)carve((size_t)N_HID * N_HID * sizeof(__bf16));
  __bf16* Wt2b = (__bf16*)carve((size_t)N_HID * N_CLS * sizeof(__bf16));

  // dynamic-LDS opt-in (host-side attribute set; graph-capture safe)
  const int shm256 = (N_FEAT * 2 + 32) * 128;   // 68 KB
  const int shm512 = (N_HID * 2 + 32) * 128;    // 132 KB
  (void)hipFuncSetAttribute((const void*)&gemm_kernel<N_FEAT, N_HID, true>,
                            hipFuncAttributeMaxDynamicSharedMemorySize, shm256);
  (void)hipFuncSetAttribute((const void*)&gemm_kernel<N_HID, N_HID, true>,
                            hipFuncAttributeMaxDynamicSharedMemorySize, shm512);
  (void)hipFuncSetAttribute((const void*)&gemm_kernel<N_HID, N_CLS, false>,
                            hipFuncAttributeMaxDynamicSharedMemorySize, shm512);

  const int TB = 256;
  // 1) weights -> bf16, transposed to N x K
  wconv_kernel<<<(N_FEAT * N_HID + TB - 1) / TB, TB, 0, stream>>>(W1a, Wt1a, N_FEAT, N_HID);
  wconv_kernel<<<(N_HID * N_HID + TB - 1) / TB, TB, 0, stream>>>(W1b, Wt1b, N_HID, N_HID);
  wconv_kernel<<<(N_HID * N_HID + TB - 1) / TB, TB, 0, stream>>>(W2a, Wt2a, N_HID, N_HID);
  wconv_kernel<<<(N_HID * N_CLS + TB - 1) / TB, TB, 0, stream>>>(W2b, Wt2b, N_HID, N_CLS);

  // 2) GIN layer 1 aggregation: bufA = (1+eps1)*x + scatter_sum(x)
  {
    long long n = (long long)M * N_FEAT;
    scale_kernel<<<(unsigned)((n + TB - 1) / TB), TB, 0, stream>>>(x, eps1, bufA, n);
    long long t = (long long)E * (N_FEAT / 4);
    scatter_kernel<N_FEAT><<<(unsigned)((t + TB - 1) / TB), TB, 0, stream>>>(x, ei, bufA, E);
  }

  // 3) MLP1: bufB = relu(bufA @ W1a + b1a); bufA = relu(bufB @ W1b + b1b)
  {
    dim3 g1((M + 255) / 256, N_HID / 128);
    gemm_kernel<N_FEAT, N_HID, true><<<g1, TB, shm256, stream>>>(bufA, Wt1a, b1a, bufB, M);
    gemm_kernel<N_HID, N_HID, true><<<g1, TB, shm512, stream>>>(bufB, Wt1b, b1b, bufA, M);
  }

  // 4) GIN layer 2 aggregation: bufB = (1+eps2)*bufA + scatter_sum(bufA)
  {
    long long n = (long long)M * N_HID;
    scale_kernel<<<(unsigned)((n + TB - 1) / TB), TB, 0, stream>>>(bufA, eps2, bufB, n);
    long long t = (long long)E * (N_HID / 4);
    scatter_kernel<N_HID><<<(unsigned)((t + TB - 1) / TB), TB, 0, stream>>>(bufA, ei, bufB, E);
  }

  // 5) MLP2: bufA = relu(bufB @ W2a + b2a); out = bufA @ W2b + b2b
  {
    dim3 g1((M + 255) / 256, N_HID / 128);
    gemm_kernel<N_HID, N_HID, true><<<g1, TB, shm512, stream>>>(bufB, Wt2a, b2a, bufA, M);
    dim3 g2((M + 255) / 256, N_CLS / 128);
    gemm_kernel<N_HID, N_CLS, false><<<g2, TB, shm512, stream>>>(bufA, Wt2b, b2b, (float*)d_out, M);
  }
}
